// QLinear_60284160967070
// MI455X (gfx1250) — 8-bit matrix operand order repaired and checked by reading, compiled, not run on hardware
//
#include <hip/hip_runtime.h>
#include <stdint.h>

typedef __attribute__((ext_vector_type(8))) int v8i;

#define M_DIM 8192
#define K_DIM 4096
#define N_DIM 4096

#define INPUT_ZP   (-3)
#define OUTPUT_ZP  (-5)
#define INPUT_SCALE  0.05f
#define OUTPUT_SCALE 0.1f
#define QMIN (-128)
#define QMAX (127)

#define BK 64
#define NK (K_DIM / BK)
#define LDS_STRIDE 80
#define A_TILE_BYTES (128 * LDS_STRIDE)
#define B_TILE_BYTES (128 * LDS_STRIDE)
#define BUF_BYTES (A_TILE_BYTES + B_TILE_BYTES)
#define SMEM_BYTES (2 * BUF_BYTES)

#define ASYNC_LDS_B128(ldsaddr, voff, sbase)                                   \
    asm volatile("global_load_async_to_lds_b128 %0, %1, %2"                    \
                 :: "v"(ldsaddr), "v"(voff), "s"(sbase) : "memory")
#define WAIT_ASYNC() asm volatile("s_wait_asynccnt 0x0" ::: "memory")

__global__ void pack_a_kernel(const int* __restrict__ qin,
                              unsigned int* __restrict__ pa) {
    int idx = blockIdx.x * blockDim.x + threadIdx.x;
    int4 v = ((const int4*)qin)[idx];
    unsigned int p = (v.x & 0xff) | ((v.y & 0xff) << 8) |
                     ((v.z & 0xff) << 16) | ((unsigned int)v.w << 24);
    pa[idx] = p;
}

__global__ void pack_w_kernel(const int* __restrict__ qw,
                              const int* __restrict__ qbias,
                              const float* __restrict__ wscale,
                              unsigned int* __restrict__ pw,
                              int* __restrict__ sqbias,
                              int* __restrict__ iscale,
                              int* __restrict__ fbits) {
    __shared__ int red[256];
    const int n = blockIdx.x;
    const int t = threadIdx.x;
    const int* row = qw + (size_t)n * K_DIM;
    int s = 0;
#pragma unroll
    for (int jj = 0; jj < 4; ++jj) {
        int4 v = ((const int4*)row)[jj * 256 + t];
        unsigned int p = (v.x & 0xff) | ((v.y & 0xff) << 8) |
                         ((v.z & 0xff) << 16) | ((unsigned int)v.w << 24);
        pw[(size_t)n * (K_DIM / 4) + jj * 256 + t] = p;
        s += v.x + v.y + v.z + v.w;
    }
    red[t] = s;
    __syncthreads();
#pragma unroll
    for (int off = 128; off > 0; off >>= 1) {
        if (t < off) red[t] += red[t + off];
        __syncthreads();
    }
    if (t == 0) {
        int rowsum = red[0];
        sqbias[n] = qbias[n] - rowsum * INPUT_ZP;
        float folded = INPUT_SCALE * wscale[n] / OUTPUT_SCALE;
        int fb = 7 - (int)ceilf(log2f(folded));
        int isc = (int)rintf(folded * exp2f((float)fb));
        iscale[n] = isc;
        fbits[n]  = fb;
    }
}

extern __shared__ char smem[];

__global__ __launch_bounds__(256)
void qgemm_wmma_kernel(const signed char* __restrict__ pa,
                       const signed char* __restrict__ pw,
                       const int* __restrict__ sqbias,
                       const int* __restrict__ iscale,
                       const int* __restrict__ fbits,
                       float* __restrict__ out) {
    const int t    = threadIdx.x;
    const int lane = t & 31;
    const int wid  = t >> 5;
    const int l16  = lane & 15;
    const int hi   = lane >> 4;
    const int waveM = wid & 3;
    const int waveN = wid >> 2;

    const int mBase = blockIdx.y * 128;
    const int nBase = blockIdx.x * 128;

    const unsigned ldsBase = (unsigned)__builtin_amdgcn_groupstaticsize();

    const int ar0 = t >> 2,          ac0 = t & 3;
    const int ar1 = (t + 256) >> 2,  ac1 = (t + 256) & 3;

    const unsigned aSlot0 = (unsigned)(ar0 * LDS_STRIDE + ac0 * 16);
    const unsigned aSlot1 = (unsigned)(ar1 * LDS_STRIDE + ac1 * 16);

    const unsigned voffA0 = (unsigned)(mBase + ar0) * K_DIM + ac0 * 16;
    const unsigned voffA1 = (unsigned)(mBase + ar1) * K_DIM + ac1 * 16;
    const unsigned voffB0 = (unsigned)(nBase + ar0) * K_DIM + ac0 * 16;
    const unsigned voffB1 = (unsigned)(nBase + ar1) * K_DIM + ac1 * 16;

    const unsigned long long baseA = (unsigned long long)pa;
    const unsigned long long baseB = (unsigned long long)pw;

    ASYNC_LDS_B128(ldsBase + aSlot0,               voffA0, baseA);
    ASYNC_LDS_B128(ldsBase + aSlot1,               voffA1, baseA);
    ASYNC_LDS_B128(ldsBase + A_TILE_BYTES + aSlot0, voffB0, baseB);
    ASYNC_LDS_B128(ldsBase + A_TILE_BYTES + aSlot1, voffB1, baseB);
    WAIT_ASYNC();
    __syncthreads();

    v8i acc[2][4];
    const v8i vzero = {0, 0, 0, 0, 0, 0, 0, 0};
#pragma unroll
    for (int i = 0; i < 2; ++i)
#pragma unroll
        for (int j = 0; j < 4; ++j) acc[i][j] = vzero;

    int aRow[2];
#pragma unroll
    for (int tm = 0; tm < 2; ++tm)
        aRow[tm] = (waveM * 32 + tm * 16 + l16) * LDS_STRIDE + hi * 16;
    int bRow[4];
#pragma unroll
    for (int tn = 0; tn < 4; ++tn)
        bRow[tn] = A_TILE_BYTES + (waveN * 64 + tn * 16 + l16) * LDS_STRIDE + hi * 16;

    unsigned curOff = 0;

#pragma unroll 1
    for (int kt = 0; kt < NK; ++kt) {
        if (kt + 1 < NK) {
            const unsigned nxt = curOff ^ BUF_BYTES;
            const unsigned long long kb = (unsigned long long)(kt + 1) * BK;
            ASYNC_LDS_B128(ldsBase + nxt + aSlot0,                voffA0, baseA + kb);
            ASYNC_LDS_B128(ldsBase + nxt + aSlot1,                voffA1, baseA + kb);
            ASYNC_LDS_B128(ldsBase + nxt + A_TILE_BYTES + aSlot0, voffB0, baseB + kb);
            ASYNC_LDS_B128(ldsBase + nxt + A_TILE_BYTES + aSlot1, voffB1, baseB + kb);
        }

        const char* buf = smem + curOff;
        v8i afrag[2];
#pragma unroll
        for (int tm = 0; tm < 2; ++tm) {
            const char* ap = buf + aRow[tm];
#pragma unroll
            for (int c = 0; c < 4; ++c)
                ((int64_t*)&afrag[tm])[c] = *(const int64_t*)(ap + (c >> 1) * 32 + (c & 1) * 8);
        }
        v8i bfrag[4];
#pragma unroll
        for (int tn = 0; tn < 4; ++tn) {
            const char* bp = buf + bRow[tn];
            ((int4*)&bfrag[tn])[0] = *(const int4*)(bp);
            ((int4*)&bfrag[tn])[1] = *(const int4*)(bp + 32);
        }
#pragma unroll
        for (int tm = 0; tm < 2; ++tm)
#pragma unroll
            for (int tn = 0; tn < 4; ++tn)
                acc[tm][tn] = __builtin_amdgcn_wmma_i32_16x16x64_iu8(
                     true, afrag[tm],
                     true, bfrag[tn],
                    acc[tm][tn],
                     false,  false);

        WAIT_ASYNC();
        __syncthreads();
        curOff ^= BUF_BYTES;
    }

#pragma unroll
    for (int tn = 0; tn < 4; ++tn) {
        const int gn = nBase + waveN * 64 + tn * 16 + l16;
        const int     sq  = sqbias[gn];
        const int64_t isc = (int64_t)iscale[gn];
        const int     fb  = fbits[gn];
#pragma unroll
        for (int tm = 0; tm < 2; ++tm) {
            const int gmb = mBase + waveM * 32 + tm * 16 + hi * 8;
            v8i c = acc[tm][tn];
#pragma unroll
            for (int r = 0; r < 8; ++r) {
                int64_t p = (int64_t)(c[r] + sq) * isc;
                int64_t s = (p >> fb) + OUTPUT_ZP;
                s = s < QMIN ? QMIN : (s > QMAX ? QMAX : s);
                out[(size_t)(gmb + r) * N_DIM + gn] = (float)(int)s;
            }
        }
    }
}

extern "C" void kernel_launch(void* const* d_in, const int* in_sizes, int n_in,
                              void* d_out, int out_size, void* d_ws, size_t ws_size,
                              hipStream_t stream) {
    const int*   qinput = (const int*)d_in[0];
    const int*   qweight = (const int*)d_in[1];
    const int*   qbias = (const int*)d_in[2];
    const float* wscale = (const float*)d_in[3];
    float* out = (float*)d_out;

    signed char* pa = (signed char*)d_ws;
    signed char* pw = pa + (size_t)M_DIM * K_DIM;
    int* sqb = (int*)(pw + (size_t)N_DIM * K_DIM);
    int* isc = sqb + N_DIM;
    int* fb  = isc + N_DIM;

    pack_a_kernel<<<(M_DIM * (size_t)K_DIM / 4) / 256, 256, 0, stream>>>(
        qinput, (unsigned int*)pa);

    pack_w_kernel<<<N_DIM, 256, 0, stream>>>(
        qweight, qbias, wscale, (unsigned int*)pw, sqb, isc, fb);

    dim3 grid(N_DIM / 128, M_DIM / 128);
    qgemm_wmma_kernel<<<grid, 256, SMEM_BYTES, stream>>>(pa, pw, sqb, isc, fb, out);
}
